// WindowAttention_84241488543946
// MI455X (gfx1250) — matrix operand order repaired and checked by reading, not compiled, not run on hardware
//
#include <hip/hip_runtime.h>

// ---------------------------------------------------------------------------
// MI455X fused windowed attention (gfx1250, wave32, WMMA bf16).
//   x:[4,16,2048,512]f32 -> qkv GEMM -> 8-head window(64) attention -> proj GEMM
// One workgroup per 64-token window (256 thr = 8 wave32, wave h owns head h).
// Weights pre-converted to transposed bf16 in d_ws (~2 MB, L2-resident).
// Loop order nt->ks->mt shares each global B fragment across 4 WMMAs and gives
// 4 independent accumulation chains for XDL latency hiding.
// ---------------------------------------------------------------------------

typedef __attribute__((ext_vector_type(16))) __bf16 v16bf;
typedef __attribute__((ext_vector_type(8)))  float  v8f;

#define D_MODEL 512
#define NQKV    1536
#define W       64
#define HD      64
#define LDK     520   // row stride (bf16) for x / Q / K(P) / O buffers (bank-conflict pad)
#define VTLD    72    // row stride (bf16) for V-transposed buffer

__device__ __forceinline__ unsigned short f2bf(float f) {
  union { float f; unsigned u; } v; v.f = f;
  unsigned r = v.u + 0x7FFFu + ((v.u >> 16) & 1u);   // round-to-nearest-even
  return (unsigned short)(r >> 16);
}

// pack two f32 -> packed bf16 dword (v_cvt_pk_bf16_f32 when available)
__device__ __forceinline__ unsigned pack_bf16(float a, float b) {
#if __has_builtin(__builtin_amdgcn_cvt_pk_bf16_f32)
  typedef __attribute__((ext_vector_type(2))) __bf16 v2bf;
  union { v2bf v; unsigned u; } c;
  c.v = __builtin_amdgcn_cvt_pk_bf16_f32(a, b);
  return c.u;
#else
  return (unsigned)f2bf(a) | ((unsigned)f2bf(b) << 16);
#endif
}

union FragBF { uint4 u4[2]; v16bf v; };

// A-operand 16x32 bf16 fragment from row-major [row][k] storage, leading dim ld.
// lanes 0-15: rows, K = base+0..7 and base+16..23 ; lanes 16-31: K = +8..15 / +24..31
__device__ __forceinline__ v16bf frag_a(const __bf16* base, int row, int k, int ld) {
  const int lane = threadIdx.x & 31;
  const int r  = row + (lane & 15);
  const int kk = k + ((lane >> 4) << 3);
  FragBF f;
  f.u4[0] = *(const uint4*)(base + r * ld + kk);
  f.u4[1] = *(const uint4*)(base + r * ld + kk + 16);
  return f.v;
}

// B-operand 32x16 bf16 fragment. Source stored "transposed": row index = output
// column n, contiguous K in the row. lanes 0-15: K = base+0..7 and base+16..23 ; lanes 16-31: K = +8..15 / +24..31
__device__ __forceinline__ v16bf frag_b(const __bf16* base, int row, int k, int ld) {
  const int lane = threadIdx.x & 31;
  const int r  = row + (lane & 15);
  const int kk = k + ((lane >> 4) << 3);
  FragBF f;
  f.u4[0] = *(const uint4*)(base + r * ld + kk);
  f.u4[1] = *(const uint4*)(base + r * ld + kk + 16);
  return f.v;
}

__device__ __forceinline__ v8f wmma_bf16(v16bf a, v16bf b, v8f c) {
  return __builtin_amdgcn_wmma_f32_16x16x32_bf16(false, a, false, b, (short)0, c,
                                                 false, false);
}

// ---------------------------------------------------------------------------
// Weight prep: fp32 [K][N] -> bf16 transposed [N][K]
// ---------------------------------------------------------------------------
extern "C" __global__ __launch_bounds__(256)
void wattn_convert_weights(const float* __restrict__ wqkv,
                           const float* __restrict__ wproj,
                           __bf16* __restrict__ wqkvT,
                           __bf16* __restrict__ wprojT) {
  const int i = blockIdx.x * 256 + threadIdx.x;
  if (i < D_MODEL * NQKV) {
    const int k = i / NQKV, n = i % NQKV;
    *(unsigned short*)(wqkvT + n * D_MODEL + k) = f2bf(wqkv[i]);
  }
  if (i < D_MODEL * D_MODEL) {
    const int k = i / D_MODEL, n = i % D_MODEL;
    *(unsigned short*)(wprojT + n * D_MODEL + k) = f2bf(wproj[i]);
  }
}

// ---------------------------------------------------------------------------
// Fused per-window kernel.
// ---------------------------------------------------------------------------
extern "C" __global__ __launch_bounds__(256)
void wattn_fused(const float* __restrict__ x,
                 const float* __restrict__ b_qkv,
                 const float* __restrict__ b_proj,
                 const __bf16* __restrict__ wqkvT,
                 const __bf16* __restrict__ wprojT,
                 float* __restrict__ out) {
  extern __shared__ __align__(16) char smem[];
  __bf16* xb = (__bf16*)smem;        // 64 x LDK  : x window (bf16)
  __bf16* Qb = xb + W * LDK;         // 64 x LDK  : Q, later attention output O
  __bf16* Kb = Qb + W * LDK;         // 64 x LDK  : K, later softmax probs P
  __bf16* Vt = Kb + W * LDK;         // 512 x VTLD: V transposed [d][token]

  const int tid  = threadIdx.x;
  const int lane = tid & 31;
  const int wid  = tid >> 5;
  const long tok0 = (long)blockIdx.x * W;           // grid = B*F*nw = 2048
  const float* xg = x + tok0 * D_MODEL;

  // ---- stage x window (64x512 f32) into LDS as bf16 ----
  #pragma unroll
  for (int it = 0; it < (W * D_MODEL) / (256 * 4); ++it) {
    const int idx = (it * 256 + tid) * 4;
    const int r = idx >> 9;
    const int c = idx & (D_MODEL - 1);
    const float4 v = *(const float4*)(xg + (long)r * D_MODEL + c);
    uint2 p;
    p.x = pack_bf16(v.x, v.y);
    p.y = pack_bf16(v.z, v.w);
    *(uint2*)(xb + r * LDK + c) = p;
  }
  __syncthreads();

  // ---- GEMM1: qkv = x @ Wqkv + b_qkv; Q,K row-major, V transposed into LDS ----
  {
    const int nslab = wid * 192;                     // 8 waves x 192 cols = 1536
    for (int nt = 0; nt < 12; ++nt) {
      const int ncol  = nslab + nt * 16;
      const int nlane = ncol + (lane & 15);
      const float bias = b_qkv[nlane];
      if (nt + 1 < 12) {   // prefetch next weight slab row for this lane
        __builtin_prefetch(wqkvT + (long)(nlane + 16) * D_MODEL + ((lane >> 4) << 4),
                           0, 1);
      }
      v8f acc[4] = {};
      #pragma unroll
      for (int ks = 0; ks < 16; ++ks) {
        const v16bf b = frag_b(wqkvT, ncol, ks * 32, D_MODEL);  // one L2 fetch
        #pragma unroll
        for (int mt = 0; mt < 4; ++mt) {                        // 4 WMMAs per B
          const v16bf a = frag_a(xb, mt * 16, ks * 32, LDK);
          acc[mt] = wmma_bf16(a, b, acc[mt]);
        }
      }
      #pragma unroll
      for (int mt = 0; mt < 4; ++mt) {
        const int m0 = mt * 16 + ((lane >> 4) << 3);
        if (ncol < 1024) {                           // Q or K: row-major [token][h*64+d]
          __bf16* dst = (ncol < 512) ? Qb : Kb;
          const int col = nlane & 511;
          #pragma unroll
          for (int j = 0; j < 8; ++j)
            *(unsigned short*)(dst + (m0 + j) * LDK + col) = f2bf(acc[mt][j] + bias);
        } else {                                     // V: transposed [d][token], packed b128
          const int drow = nlane - 1024;
          uint4 pk;
          pk.x = pack_bf16(acc[mt][0] + bias, acc[mt][1] + bias);
          pk.y = pack_bf16(acc[mt][2] + bias, acc[mt][3] + bias);
          pk.z = pack_bf16(acc[mt][4] + bias, acc[mt][5] + bias);
          pk.w = pack_bf16(acc[mt][6] + bias, acc[mt][7] + bias);
          *(uint4*)(Vt + drow * VTLD + m0) = pk;
        }
      }
    }
  }
  __syncthreads();

  // ---- attention: wave `wid` handles head `wid` ----
  {
    const int h = wid;
    const __bf16* Qh = Qb + h * HD;
    const __bf16* Kh = Kb + h * HD;

    // S = (Q Kt): share each K fragment across the 4 q-row tiles
    v8f S[4][4] = {};
    #pragma unroll
    for (int ks = 0; ks < 2; ++ks)
      #pragma unroll
      for (int nk = 0; nk < 4; ++nk) {
        const v16bf b = frag_b(Kh, nk * 16, ks * 32, LDK);
        #pragma unroll
        for (int mq = 0; mq < 4; ++mq) {
          const v16bf a = frag_a(Qh, mq * 16, ks * 32, LDK);
          S[mq][nk] = wmma_bf16(a, b, S[mq][nk]);
        }
      }

    // softmax over k (16 lanes x 4 tiles); write P bf16 into this head's K slice
    const float SC = 0.125f * 1.44269504088896340736f;         // 1/sqrt(64) * log2(e)
    #pragma unroll
    for (int mq = 0; mq < 4; ++mq) {
      #pragma unroll
      for (int j = 0; j < 8; ++j) {
        float mx = fmaxf(fmaxf(S[mq][0][j], S[mq][1][j]),
                         fmaxf(S[mq][2][j], S[mq][3][j]));
        #pragma unroll
        for (int msk = 1; msk < 16; msk <<= 1)
          mx = fmaxf(mx, __shfl_xor(mx, msk, 32));
        const float e0 = exp2f((S[mq][0][j] - mx) * SC);
        const float e1 = exp2f((S[mq][1][j] - mx) * SC);
        const float e2 = exp2f((S[mq][2][j] - mx) * SC);
        const float e3 = exp2f((S[mq][3][j] - mx) * SC);
        float sum = (e0 + e1) + (e2 + e3);
        #pragma unroll
        for (int msk = 1; msk < 16; msk <<= 1)
          sum += __shfl_xor(sum, msk, 32);
        const float r = __builtin_amdgcn_rcpf(sum);
        const int q = mq * 16 + ((lane >> 4) << 3) + j;
        __bf16* prow = Kb + q * LDK + h * HD + (lane & 15);
        *(unsigned short*)(prow +  0) = f2bf(e0 * r);
        *(unsigned short*)(prow + 16) = f2bf(e1 * r);
        *(unsigned short*)(prow + 32) = f2bf(e2 * r);
        *(unsigned short*)(prow + 48) = f2bf(e3 * r);
      }
    }

    // O = P @ V ; share each V fragment across the 4 q-row tiles
    #pragma unroll
    for (int nd = 0; nd < 4; ++nd) {
      v8f accO[4] = {};
      #pragma unroll
      for (int ks = 0; ks < 2; ++ks) {
        const v16bf b = frag_b(Vt + h * HD * VTLD, nd * 16, ks * 32, VTLD); // V^T
        #pragma unroll
        for (int mq = 0; mq < 4; ++mq) {
          const v16bf a = frag_a(Kb + h * HD, mq * 16, ks * 32, LDK);       // P
          accO[mq] = wmma_bf16(a, b, accO[mq]);
        }
      }
      #pragma unroll
      for (int mq = 0; mq < 4; ++mq) {
        const int m0  = mq * 16 + ((lane >> 4) << 3);
        const int col = h * HD + nd * 16 + (lane & 15);
        #pragma unroll
        for (int j = 0; j < 8; ++j)
          *(unsigned short*)(Qb + (m0 + j) * LDK + col) = f2bf(accO[mq][j]);
      }
    }
  }
  __syncthreads();

  // ---- GEMM2: out = O @ Wproj + b_proj ----
  {
    const int nslab = wid * 64;                       // 8 waves x 64 cols = 512
    float* og = out + tok0 * D_MODEL;
    #pragma unroll
    for (int nt = 0; nt < 4; ++nt) {
      const int ncol  = nslab + nt * 16;
      const int nlane = ncol + (lane & 15);
      const float bias = b_proj[nlane];
      if (nt + 1 < 4) {
        __builtin_prefetch(wprojT + (long)(nlane + 16) * D_MODEL + ((lane >> 4) << 4),
                           0, 1);
      }
      v8f acc[4] = {};
      #pragma unroll
      for (int ks = 0; ks < 16; ++ks) {
        const v16bf b = frag_b(wprojT, ncol, ks * 32, D_MODEL);
        #pragma unroll
        for (int mt = 0; mt < 4; ++mt) {
          const v16bf a = frag_a(Qb, mt * 16, ks * 32, LDK);
          acc[mt] = wmma_bf16(a, b, acc[mt]);
        }
      }
      #pragma unroll
      for (int mt = 0; mt < 4; ++mt) {
        const int m0 = mt * 16 + ((lane >> 4) << 3);
        #pragma unroll
        for (int j = 0; j < 8; ++j)
          og[(long)(m0 + j) * D_MODEL + nlane] = acc[mt][j] + bias;
      }
    }
  }
}

// ---------------------------------------------------------------------------
extern "C" void kernel_launch(void* const* d_in, const int* in_sizes, int n_in,
                              void* d_out, int out_size, void* d_ws, size_t ws_size,
                              hipStream_t stream) {
  const float* x      = (const float*)d_in[0];
  const float* w_qkv  = (const float*)d_in[1];
  const float* b_qkv  = (const float*)d_in[2];
  const float* w_proj = (const float*)d_in[3];
  const float* b_proj = (const float*)d_in[4];

  __bf16* wqkvT  = (__bf16*)d_ws;                    // [1536][512] bf16
  __bf16* wprojT = wqkvT + (size_t)D_MODEL * NQKV;   // [512][512]  bf16

  const int nconv = D_MODEL * NQKV;
  wattn_convert_weights<<<(nconv + 255) / 256, 256, 0, stream>>>(w_qkv, w_proj,
                                                                 wqkvT, wprojT);

  const size_t lds_bytes =
      (size_t)(3 * W * LDK + D_MODEL * VTLD) * sizeof(unsigned short); // 273,408 B
  hipFuncSetAttribute((const void*)wattn_fused,
                      hipFuncAttributeMaxDynamicSharedMemorySize, (int)lds_bytes);

  const int nwindows = 4 * 16 * (2048 / W);          // B*F*nw = 2048
  wattn_fused<<<nwindows, 256, lds_bytes, stream>>>(x, b_qkv, b_proj, wqkvT,
                                                    wprojT, (float*)d_out);
}
